// EncoderLayer_14379550507719
// MI455X (gfx1250) — matrix operand order repaired and checked by reading, not compiled, not run on hardware
//
#include <hip/hip_runtime.h>

// ---------------------------------------------------------------------------
// Problem constants (fixed by the reference)
// ---------------------------------------------------------------------------
#define Bc   4
#define Sc   2048
#define Dc   1024
#define Hc   16
#define DKc  64
#define DVc  64
#define DFFc 4096
#define Mc   (Bc * Sc)          // 8192 rows of activations
#define EPSc 1e-5f

typedef unsigned short u16;
typedef __attribute__((ext_vector_type(16))) __bf16        v16bf;
typedef __attribute__((ext_vector_type(8)))  float         v8f;
typedef __attribute__((ext_vector_type(4)))  unsigned int  v4u;

// Fragment: 16 bf16 = 8 VGPRs = two 128-bit loads
union FragBF {
    v16bf v;
    v4u   q[2];
    u16   u[16];
};

// fp32 -> bf16 RNE (conversion passes, epilogues and in-loop attn A conversion)
__device__ __forceinline__ u16 f2bf(float x) {
    unsigned int u = __float_as_uint(x);
    u += 0x7fffu + ((u >> 16) & 1u);
    return (u16)(u >> 16);
}
__device__ __forceinline__ void put_pair(FragBF& f, int v, float x, float y) {
    f.u[2 * v]     = f2bf(x);
    f.u[2 * v + 1] = f2bf(y);
}
// A-fragment K offset for 16-bit A 16x32 (per ISA 7.12.2), context kernel only
__device__ __forceinline__ int ka_off(int v, int half) {
    return ((v < 4) ? (2 * v) : (16 + 2 * (v - 4))) + 8 * half;
}

// ---------------------------------------------------------------------------
// One-time conversion passes (each element converted exactly once per call)
// ---------------------------------------------------------------------------
__global__ void __launch_bounds__(256)
cvt_f32_bf16(const float* __restrict__ src, u16* __restrict__ dst, int n)
{
    const int i = blockIdx.x * 256 + threadIdx.x;
    if (i < n) dst[i] = f2bf(src[i]);
}

// W[K,N] f32  ->  Wt[N,K] bf16  (coalesced read, scattered b16 write; small)
__global__ void __launch_bounds__(256)
cvt_transpose_w(const float* __restrict__ W, u16* __restrict__ Wt, int K, int N)
{
    const int i = blockIdx.x * 256 + threadIdx.x;
    if (i < K * N) {
        const int k = i / N, n = i % N;
        Wt[(size_t)n * K + k] = f2bf(W[i]);
    }
}

// V bf16 [B,S,H*DV] -> VT bf16 [B,H,DV,S] (per-head transpose for context B)
__global__ void __launch_bounds__(256)
transpose_v(const u16* __restrict__ V, u16* __restrict__ VT)
{
    const int i = blockIdx.x * 256 + threadIdx.x;   // coalesced read index
    const int d = i & (DVc - 1);
    const int h = (i >> 6) & (Hc - 1);
    const int s = (i >> 10) & (Sc - 1);
    const int b = i >> 21;
    VT[(((size_t)(b * Hc + h) * DVc) + d) * Sc + s] = V[i];
}

// ---------------------------------------------------------------------------
// GEMM: C[M,N] = epilogue(Abf[M,K] @ Wt[N,K]^T), all-bf16 operands.
// One wave = one 16x16 tile; inner loop = 4x global_load_b128 + 1 WMMA.
// Weights (<=8MB bf16) and activations fit the 192MB L2 -> no LDS blocking.
// ---------------------------------------------------------------------------
__global__ void __launch_bounds__(256)
gemm_bf16_wmma(const u16* __restrict__ A, const u16* __restrict__ Wt,
               float* __restrict__ Cf, u16* __restrict__ Cb,
               const float* __restrict__ resid,
               int M, int N, int K, int do_relu)
{
    const int lane = threadIdx.x & 31;
    const int wv   = threadIdx.x >> 5;
    const long gw  = (long)blockIdx.x * (blockDim.x >> 5) + wv;
    const int  ntn = N >> 4;
    const long tiles = (long)(M >> 4) * ntn;
    if (gw >= tiles) return;                      // wave-uniform: EXEC all-1

    const int tm   = (int)(gw / ntn);
    const int tn   = (int)(gw % ntn);
    const int half = lane >> 4;
    const int l16  = lane & 15;
    const int bcol = tn * 16 + l16;
    const u16* Arow = A  + (size_t)(tm * 16 + l16) * K;
    const u16* Brow = Wt + (size_t)bcol * K;

    v8f acc = {};
#pragma unroll 2
    for (int k0 = 0; k0 < K; k0 += 32) {
        __builtin_prefetch(Brow + k0 + 64, 0, 1); // speculative, non-faulting
        FragBF a, b;
        a.q[0] = *(const v4u*)(Arow + k0 + 8 * half);        // k = k0+8h..+7
        a.q[1] = *(const v4u*)(Arow + k0 + 16 + 8 * half);   // k = k0+16+8h..+7
        b.q[0] = *(const v4u*)(Brow + k0 + 8 * half);        // k = k0+8h..+7
        b.q[1] = *(const v4u*)(Brow + k0 + 16 + 8 * half);   // k = k0+16+8h..+7
        acc = __builtin_amdgcn_wmma_f32_16x16x32_bf16(
                  false, a.v, false, b.v, (short)0, acc, false, false);
    }
#pragma unroll
    for (int r = 0; r < 8; ++r) {
        const int row  = tm * 16 + half * 8 + r;  // C/D layout: M = 8*half + r
        const size_t i = (size_t)row * N + bcol;
        float val = acc[r];
        if (resid)   val += resid[i];
        if (do_relu) val  = val > 0.f ? val : 0.f;
        if (Cf) Cf[i] = val;
        if (Cb) Cb[i] = f2bf(val);
    }
}

// ---------------------------------------------------------------------------
// scores[b,h,q,k] = (Q_bh[q,:] . K_bh[k,:]) / 8; Q,K bf16 [B,S,H*DK].
// The WMMA K-dim is the head dim (contiguous for both operands).
// NT stores: 1.07GB tensor, read once by softmax, cannot live in L2.
// ---------------------------------------------------------------------------
__global__ void __launch_bounds__(256)
attn_scores_wmma(const u16* __restrict__ Q, const u16* __restrict__ Kt,
                 float* __restrict__ attn)
{
    const int lane = threadIdx.x & 31;
    const int wv   = threadIdx.x >> 5;
    const long gw  = (long)blockIdx.x * (blockDim.x >> 5) + wv;
    const int  tper = Sc >> 4;                    // 128
    const long tilesPerBH = (long)tper * tper;
    const long tiles = (long)Bc * Hc * tilesPerBH;
    if (gw >= tiles) return;

    const int  bh = (int)(gw / tilesPerBH);
    const long rm = gw % tilesPerBH;
    const int  tq = (int)(rm / tper);
    const int  tk = (int)(rm % tper);
    const int  b  = bh / Hc, h = bh % Hc;
    const int  half = lane >> 4;
    const int  l16  = lane & 15;
    const int  qrow = tq * 16 + l16;
    const int  kcol = tk * 16 + l16;

    const u16* Qb = Q  + ((size_t)(b * Sc + qrow) * (Hc * DKc) + h * DKc);
    const u16* Kb = Kt + ((size_t)(b * Sc + kcol) * (Hc * DKc) + h * DKc);

    v8f acc = {};
#pragma unroll
    for (int k0 = 0; k0 < DKc; k0 += 32) {
        FragBF a, bfr;
        a.q[0]   = *(const v4u*)(Qb + k0 + 8 * half);
        a.q[1]   = *(const v4u*)(Qb + k0 + 16 + 8 * half);
        bfr.q[0] = *(const v4u*)(Kb + k0 + 8 * half);
        bfr.q[1] = *(const v4u*)(Kb + k0 + 16 + 8 * half);
        acc = __builtin_amdgcn_wmma_f32_16x16x32_bf16(
                  false, a.v, false, bfr.v, (short)0, acc, false, false);
    }
    const float scale = 0.125f;                   // 1/sqrt(64)
#pragma unroll
    for (int r = 0; r < 8; ++r) {
        const int row = tq * 16 + half * 8 + r;
        __builtin_nontemporal_store(acc[r] * scale,
            attn + (((size_t)(b * Hc + h) * Sc + row) * Sc) + kcol);
    }
}

// ---------------------------------------------------------------------------
// In-place masked softmax over last dim (S=2048). One block/row, 8 elems per
// thread in registers -> exactly 1 NT read + 1 NT write of the 1.07GB tensor.
// ---------------------------------------------------------------------------
__global__ void __launch_bounds__(256)
softmax_mask(float* __restrict__ attn, const unsigned char* __restrict__ mask)
{
    const size_t row = blockIdx.x;                // 0 .. B*H*S-1
    const int b = (int)(row / ((size_t)Hc * Sc));
    const int q = (int)(row % Sc);
    float* p = attn + row * Sc;
    const unsigned char* m = mask + ((size_t)b * Sc + q) * Sc;
    const int t = threadIdx.x;

    float vals[8];
    float mx = -3.4e38f;
#pragma unroll
    for (int i = 0; i < 8; ++i) {
        const int j = t + i * 256;
        float s = __builtin_nontemporal_load(p + j);
        if (m[j]) s = -1e9f;                      // mask is small: L2-resident
        vals[i] = s;
        mx = fmaxf(mx, s);
    }
    __shared__ float red[256];
    red[t] = mx; __syncthreads();
    for (int o = 128; o > 0; o >>= 1) {
        if (t < o) red[t] = fmaxf(red[t], red[t + o]);
        __syncthreads();
    }
    mx = red[0]; __syncthreads();

    float sum = 0.f;
#pragma unroll
    for (int i = 0; i < 8; ++i) { vals[i] = __expf(vals[i] - mx); sum += vals[i]; }
    red[t] = sum; __syncthreads();
    for (int o = 128; o > 0; o >>= 1) {
        if (t < o) red[t] += red[t + o];
        __syncthreads();
    }
    const float inv = 1.f / red[0];
#pragma unroll
    for (int i = 0; i < 8; ++i)
        __builtin_nontemporal_store(vals[i] * inv, p + t + i * 256);
}

// ---------------------------------------------------------------------------
// ctx[b,q,h*64+d] = sum_k attn[b,h,q,k] * V[b,k,h*64+d].
// A = attn f32 (must stay f32 in d_out; converted in-loop, reused x4 across
// the 4 DV tiles by consecutive waves -> L0/L2 catches it).
// B = VT bf16 [B,H,DV,S]: contiguous 128-bit fragment loads.
// Output written bf16 (only consumed by the fc GEMM).
// ---------------------------------------------------------------------------
__global__ void __launch_bounds__(256)
attn_context_wmma(const float* __restrict__ attn, const u16* __restrict__ VT,
                  u16* __restrict__ ctx)
{
    const int lane = threadIdx.x & 31;
    const int wv   = threadIdx.x >> 5;
    const long gw  = (long)blockIdx.x * (blockDim.x >> 5) + wv;
    const int  tqn = Sc >> 4;                     // 128
    const int  tdn = DVc >> 4;                    // 4
    const long tilesPerBH = (long)tqn * tdn;
    const long tiles = (long)Bc * Hc * tilesPerBH;
    if (gw >= tiles) return;

    const int  bh = (int)(gw / tilesPerBH);
    const long rm = gw % tilesPerBH;
    const int  tq = (int)(rm / tdn);
    const int  td = (int)(rm % tdn);
    const int  b  = bh / Hc, h = bh % Hc;
    const int  half = lane >> 4;
    const int  l16  = lane & 15;
    const int  qrow = tq * 16 + l16;
    const int  dcol = td * 16 + l16;

    const float* Ab = attn + ((size_t)(b * Hc + h) * Sc + qrow) * Sc;
    const u16*  Vrow = VT + (((size_t)(b * Hc + h) * DVc) + dcol) * Sc;

    v8f acc = {};
    for (int k0 = 0; k0 < Sc; k0 += 32) {
        FragBF a, bfr;
#pragma unroll
        for (int v = 0; v < 8; ++v) {
            const float2 av = *(const float2*)(Ab + k0 + ka_off(v, half));
            put_pair(a, v, av.x, av.y);
        }
        bfr.q[0] = *(const v4u*)(Vrow + k0 + 8 * half);
        bfr.q[1] = *(const v4u*)(Vrow + k0 + 16 + 8 * half);
        acc = __builtin_amdgcn_wmma_f32_16x16x32_bf16(
                  false, a.v, false, bfr.v, (short)0, acc, false, false);
    }
#pragma unroll
    for (int r = 0; r < 8; ++r) {
        const int row = tq * 16 + half * 8 + r;
        ctx[(size_t)(b * Sc + row) * (Hc * DVc) + h * DVc + dcol] = f2bf(acc[r]);
    }
}

// ---------------------------------------------------------------------------
// LayerNorm over rows of D=1024. Writes f32 (residual/next-LN use) and
// optionally bf16 (next GEMM's A operand).
// ---------------------------------------------------------------------------
__global__ void __launch_bounds__(256)
layernorm_rows(const float* __restrict__ X, const float* __restrict__ g,
               const float* __restrict__ bt, float* __restrict__ Y,
               u16* __restrict__ Yb)
{
    const size_t row = blockIdx.x;
    const float* x = X + row * Dc;
    const int t = threadIdx.x;
    float v[4];
    float s = 0.f;
#pragma unroll
    for (int i = 0; i < 4; ++i) { v[i] = x[t + i * 256]; s += v[i]; }

    __shared__ float red[256];
    red[t] = s; __syncthreads();
    for (int o = 128; o > 0; o >>= 1) {
        if (t < o) red[t] += red[t + o];
        __syncthreads();
    }
    const float mu = red[0] * (1.f / Dc); __syncthreads();

    float vs = 0.f;
#pragma unroll
    for (int i = 0; i < 4; ++i) { const float d = v[i] - mu; vs += d * d; }
    red[t] = vs; __syncthreads();
    for (int o = 128; o > 0; o >>= 1) {
        if (t < o) red[t] += red[t + o];
        __syncthreads();
    }
    const float rinv = rsqrtf(red[0] * (1.f / Dc) + EPSc);
#pragma unroll
    for (int i = 0; i < 4; ++i) {
        const int c = t + i * 256;
        const float val = (v[i] - mu) * rinv * g[c] + bt[c];
        Y[row * Dc + c] = val;
        if (Yb) Yb[row * Dc + c] = f2bf(val);
    }
}

// ---------------------------------------------------------------------------
// Host-side orchestration (all async on `stream`, no allocs — graph-safe)
// ---------------------------------------------------------------------------
extern "C" void kernel_launch(void* const* d_in, const int* in_sizes, int n_in,
                              void* d_out, int out_size, void* d_ws, size_t ws_size,
                              hipStream_t stream)
{
    (void)in_sizes; (void)n_in; (void)out_size; (void)ws_size;

    const float* X      = (const float*)d_in[0];
    const float* W_Q    = (const float*)d_in[1];
    const float* W_K    = (const float*)d_in[2];
    const float* W_V    = (const float*)d_in[3];
    const float* W_fc   = (const float*)d_in[4];
    const float* ln1_g  = (const float*)d_in[5];
    const float* ln1_b  = (const float*)d_in[6];
    const float* ffn_W1 = (const float*)d_in[7];
    const float* ffn_W2 = (const float*)d_in[8];
    const float* ln2_g  = (const float*)d_in[9];
    const float* ln2_b  = (const float*)d_in[10];
    const unsigned char* mask = (const unsigned char*)d_in[11];

    float* out  = (float*)d_out;
    float* attn = out + (size_t)Bc * Sc * Dc;     // second tuple output

    // ---- workspace layout (256B-aligned slices) ----
    char*  wsb = (char*)d_ws;
    size_t off = 0;
    auto take = [&](size_t bytes) {
        void* p = wsb + off;
        off += (bytes + 255) & ~(size_t)255;
        return p;
    };
    u16*   Xbf   = (u16*)take((size_t)Mc * Dc * 2);
    u16*   WQt   = (u16*)take((size_t)Dc * Dc * 2);
    u16*   WKt   = (u16*)take((size_t)Dc * Dc * 2);
    u16*   WVt   = (u16*)take((size_t)Dc * Dc * 2);
    u16*   Wfct  = (u16*)take((size_t)Dc * Dc * 2);
    u16*   W1t   = (u16*)take((size_t)Dc * DFFc * 2);
    u16*   W2t   = (u16*)take((size_t)DFFc * Dc * 2);
    u16*   Qbf   = (u16*)take((size_t)Mc * Dc * 2);
    u16*   Kbf   = (u16*)take((size_t)Mc * Dc * 2);
    u16*   Vbf   = (u16*)take((size_t)Mc * Dc * 2);
    u16*   VTbf  = (u16*)take((size_t)Mc * Dc * 2);
    u16*   ctxbf = (u16*)take((size_t)Mc * Dc * 2);
    u16*   abf   = (u16*)take((size_t)Mc * Dc * 2);
    u16*   ffn1b = (u16*)take((size_t)Mc * DFFc * 2);
    float* tmp1  = (float*)take((size_t)Mc * Dc * 4);   // pre-LN scratch (reused)
    float* aout  = (float*)take((size_t)Mc * Dc * 4);   // attn_out f32 (residual)

    const dim3 blk(256);
    auto nblk = [](long tiles) { return (unsigned)((tiles + 7) / 8); };

    const long tilesD   = (long)(Mc / 16) * (Dc / 16);     // 32768
    const long tilesFF  = (long)(Mc / 16) * (DFFc / 16);   // 131072
    const long tilesSc  = (long)Bc * Hc * (Sc / 16) * (Sc / 16);
    const long tilesCtx = (long)Bc * Hc * (Sc / 16) * (DVc / 16);

    // 0) one-time conversions: activations row-major, weights transposed [N,K]
    cvt_f32_bf16   <<<(Mc * Dc) / 256, blk, 0, stream>>>(X, Xbf, Mc * Dc);
    cvt_transpose_w<<<(Dc * Dc) / 256, blk, 0, stream>>>(W_Q,  WQt,  Dc, Dc);
    cvt_transpose_w<<<(Dc * Dc) / 256, blk, 0, stream>>>(W_K,  WKt,  Dc, Dc);
    cvt_transpose_w<<<(Dc * Dc) / 256, blk, 0, stream>>>(W_V,  WVt,  Dc, Dc);
    cvt_transpose_w<<<(Dc * Dc) / 256, blk, 0, stream>>>(W_fc, Wfct, Dc, Dc);
    cvt_transpose_w<<<(Dc * DFFc) / 256, blk, 0, stream>>>(ffn_W1, W1t, Dc, DFFc);
    cvt_transpose_w<<<(DFFc * Dc) / 256, blk, 0, stream>>>(ffn_W2, W2t, DFFc, Dc);

    // 1) Q, K, V projections (bf16 outputs; only consumed by matrix ops)
    gemm_bf16_wmma<<<nblk(tilesD), blk, 0, stream>>>(Xbf, WQt, nullptr, Qbf, nullptr, Mc, Dc, Dc, 0);
    gemm_bf16_wmma<<<nblk(tilesD), blk, 0, stream>>>(Xbf, WKt, nullptr, Kbf, nullptr, Mc, Dc, Dc, 0);
    gemm_bf16_wmma<<<nblk(tilesD), blk, 0, stream>>>(Xbf, WVt, nullptr, Vbf, nullptr, Mc, Dc, Dc, 0);
    transpose_v<<<(Mc * Dc) / 256, blk, 0, stream>>>(Vbf, VTbf);

    // 2) scores = QK^T / 8 -> attn region of d_out (NT)
    attn_scores_wmma<<<nblk(tilesSc), blk, 0, stream>>>(Qbf, Kbf, attn);

    // 3) masked softmax in place (NT)
    softmax_mask<<<(unsigned)((size_t)Bc * Hc * Sc), blk, 0, stream>>>(attn, mask);

    // 4) context = attn @ V -> bf16
    attn_context_wmma<<<nblk(tilesCtx), blk, 0, stream>>>(attn, VTbf, ctxbf);

    // 5) attn_out = LN1(context @ W_fc + X); f32 + bf16 copies
    gemm_bf16_wmma<<<nblk(tilesD), blk, 0, stream>>>(ctxbf, Wfct, tmp1, nullptr, X, Mc, Dc, Dc, 0);
    layernorm_rows<<<(unsigned)Mc, blk, 0, stream>>>(tmp1, ln1_g, ln1_b, aout, abf);

    // 6) ffn1 = relu(attn_out @ W1) -> bf16
    gemm_bf16_wmma<<<nblk(tilesFF), blk, 0, stream>>>(abf, W1t, nullptr, ffn1b, nullptr, Mc, DFFc, Dc, 1);

    // 7) enc_outputs = LN2(ffn1 @ W2 + attn_out)
    gemm_bf16_wmma<<<nblk(tilesD), blk, 0, stream>>>(ffn1b, W2t, tmp1, nullptr, aout, Mc, Dc, DFFc, 0);
    layernorm_rows<<<(unsigned)Mc, blk, 0, stream>>>(tmp1, ln2_g, ln2_b, out, nullptr);
}
